// SeqSelfAttention_37503654428810
// MI455X (gfx1250) — matrix operand order repaired and checked by reading, not compiled, not run on hardware
//
#include <hip/hip_runtime.h>

typedef __attribute__((ext_vector_type(16))) __bf16 v16bf;
typedef __attribute__((ext_vector_type(8)))  __bf16 bf16x8;
typedef __attribute__((ext_vector_type(8)))  float  v8f;

#if __has_builtin(__builtin_amdgcn_tanhf)
#define TANHF(v) __builtin_amdgcn_tanhf(v)
#else
#define TANHF(v) tanhf(v)
#endif

namespace {
constexpr int Bb = 4, Ll = 1024, Dd = 128, Uu = 32;
constexpr int AP = 1032;  // padded bf16 alpha row stride (bank-conflict-free A frags)
}

// ---------------- Kernel 1: q = x@Wt^T ; k' = x@Wx^T + bh ----------------
__global__ __launch_bounds__(256) void qk_kernel(
    const float* __restrict__ x, const float* __restrict__ Wt,
    const float* __restrict__ Wx, const float* __restrict__ bh,
    float* __restrict__ qp, float* __restrict__ kp)
{
  __shared__ __align__(16) float xrow[4][Dd];
  const int t  = threadIdx.x;
  const int rl = t >> 6;        // local row 0..3
  const int tr = t & 63;
  const long row = (long)blockIdx.x * 4 + rl;   // b*L + s
  ((float2*)xrow[rl])[tr] = ((const float2*)(x + row * Dd))[tr];
  __syncthreads();
  const int u = tr & 31;
  const int which = tr >> 5;    // 0 -> q(Wt), 1 -> k'(Wx)+bh
  const float4* w4 = (const float4*)((which ? Wx : Wt) + u * Dd);
  const float4* x4 = (const float4*)xrow[rl];
  float acc = 0.f;
#pragma unroll
  for (int c = 0; c < Dd / 4; ++c) {
    float4 wv = w4[c], xv = x4[c];
    acc = fmaf(wv.x, xv.x, acc);
    acc = fmaf(wv.y, xv.y, acc);
    acc = fmaf(wv.z, xv.z, acc);
    acc = fmaf(wv.w, xv.w, acc);
  }
  if (which) kp[row * Uu + u] = acc + bh[u];
  else       qp[row * Uu + u] = acc;
}

// ---------------- Kernel 2: xT[b][d][j] = bf16(x[b][j][d]) ----------------
__global__ __launch_bounds__(256) void xt_kernel(const float* __restrict__ x,
                                                 __bf16* __restrict__ xT)
{
  __shared__ float tile[32][33];
  const int bid = blockIdx.x;
  const int b   = bid >> 7;
  const int rem = bid & 127;
  const int j0  = (rem >> 2) << 5;
  const int d0  = (rem & 3) << 5;
  const int tx  = threadIdx.x & 31;
  const int ty  = threadIdx.x >> 5;  // 0..7
#pragma unroll
  for (int i = 0; i < 4; ++i) {
    int j = j0 + ty + i * 8;
    tile[ty + i * 8][tx] = x[((long)b * Ll + j) * Dd + d0 + tx];
  }
  __syncthreads();
#pragma unroll
  for (int i = 0; i < 4; ++i) {
    int d = d0 + ty + i * 8;
    xT[((long)b * Dd + d) * Ll + j0 + tx] = (__bf16)tile[tx][ty + i * 8];
  }
}

// ------- Kernel 3: fused scores + softmax + context GEMM (WMMA bf16) -------
__global__ __launch_bounds__(256) void attn_kernel(
    const __bf16* __restrict__ xT, const float* __restrict__ qp,
    const float* __restrict__ kp, const float* __restrict__ Wa,
    const float* __restrict__ ba, float* __restrict__ out)
{
  __shared__ __align__(16) __bf16 alphaB[16 * AP];  // 33 KB score tile
  __shared__ __align__(16) float  qt[16][Uu];
  __shared__ __align__(16) float  wa_s[Uu];
  __shared__ float inv_s[16];

  const int t  = threadIdx.x;
  const int b  = blockIdx.x >> 6;
  const int s0 = (blockIdx.x & 63) << 4;
  const long rowbase = (long)b * Ll + s0;

  for (int idx = t; idx < 16 * Uu; idx += 256)
    qt[idx >> 5][idx & 31] = qp[(rowbase + (idx >> 5)) * Uu + (idx & 31)];
  if (t < Uu) wa_s[t] = Wa[t];
  __syncthreads();

  // ---- Phase A: scores alpha[s][j] = tanh(sum_u Wa[u]*tanh(q+k') + ba) ----
  const float bav = ba[0];
  const int s   = t >> 4;   // 0..15
  const int l16 = t & 15;
  float4 qr[8];
  {
    const float4* q4 = (const float4*)qt[s];
#pragma unroll
    for (int c = 0; c < 8; ++c) qr[c] = q4[c];
  }
  const float4* wa4 = (const float4*)wa_s;
  const float4* kb4 = (const float4*)(kp + (long)b * Ll * Uu);
  __bf16* arow = alphaB + s * AP;
  float mx = -3.402823466e38f;
  for (int m = 0; m < Ll / 16; ++m) {
    const int j = l16 + (m << 4);
    const float4* kr = kb4 + j * (Uu / 4);
    float acc = 0.f;
#pragma unroll
    for (int c = 0; c < 8; ++c) {
      float4 kv = kr[c], wv = wa4[c], qv = qr[c];
      acc += wv.x * TANHF(qv.x + kv.x);
      acc += wv.y * TANHF(qv.y + kv.y);
      acc += wv.z * TANHF(qv.z + kv.z);
      acc += wv.w * TANHF(qv.w + kv.w);
    }
    float a = TANHF(acc + bav);
    arow[j] = (__bf16)a;
    mx = fmaxf(mx, a);
  }
  // ---- softmax over j (16 lanes own one s-row) ----
#pragma unroll
  for (int off = 8; off; off >>= 1) mx = fmaxf(mx, __shfl_xor(mx, off, 16));
  float sum = 0.f;
  for (int m = 0; m < Ll / 16; ++m) {
    const int j = l16 + (m << 4);
    float e = __expf((float)arow[j] - mx);
    arow[j] = (__bf16)e;          // unnormalized; 1/sum folded into C store
    sum += e;
  }
#pragma unroll
  for (int off = 8; off; off >>= 1) sum += __shfl_xor(sum, off, 16);
  if (l16 == 0) inv_s[s] = 1.0f / sum;
  __syncthreads();

  // ---- Phase B: C(16x128) = alpha(16x1024) @ x(1024x128) via WMMA bf16 ----
  const int lane = t & 31;
  const int hi   = lane >> 4;          // half-wave select
  const int n    = lane & 15;          // N column within tile / A row M
  const int d0   = (t >> 5) << 4;      // wave id * 16 columns
  const __bf16* arl  = alphaB + (lane & 15) * AP;
  const int hiA = hi * 16;
  const __bf16* bcol = xT + ((long)b * Dd + d0 + n) * Ll + hi * 16;
  v8f acc = {0.f, 0.f, 0.f, 0.f, 0.f, 0.f, 0.f, 0.f};
  union Uv { v16bf v; bf16x8 h[2]; };
#pragma unroll 4
  for (int k = 0; k < Ll; k += 32) {
    Uv a, bfr;
    // A frag, this code's contiguous k order: lanes0-15 K[k,k+16); lanes16-31 K[k+16,k+32)
    a.h[0] = *(const bf16x8*)(arl + k + hiA);
    a.h[1] = *(const bf16x8*)(arl + k + 8 + hiA);
    // B frag, same contiguous k order as A: lanes0-15 K[k,k+16); lanes16-31 K[k+16,k+32)
    bfr.h[0] = *(const bf16x8*)(bcol + k);
    bfr.h[1] = *(const bf16x8*)(bcol + k + 8);
    acc = __builtin_amdgcn_wmma_f32_16x16x32_bf16(
        false, a.v, false, bfr.v, (short)0, acc, false, false);
  }
  const long obase = rowbase * Dd + d0 + n;
#pragma unroll
  for (int r = 0; r < 8; ++r) {
    const int M = r + hi * 8;
    out[obase + (long)M * Dd] = acc[r] * inv_s[M];
  }
}

extern "C" void kernel_launch(void* const* d_in, const int* in_sizes, int n_in,
                              void* d_out, int out_size, void* d_ws, size_t ws_size,
                              hipStream_t stream) {
  (void)in_sizes; (void)n_in; (void)out_size; (void)ws_size;
  const float* x  = (const float*)d_in[0];
  const float* Wt = (const float*)d_in[1];
  const float* Wx = (const float*)d_in[2];
  const float* bh = (const float*)d_in[3];
  const float* Wa = (const float*)d_in[4];
  const float* ba = (const float*)d_in[5];
  float* out = (float*)d_out;

  float*  qp = (float*)d_ws;                         // B*L*U f32 (512 KB)
  float*  kp = qp + (size_t)Bb * Ll * Uu;            // B*L*U f32 (512 KB)
  __bf16* xT = (__bf16*)(kp + (size_t)Bb * Ll * Uu); // B*D*L bf16 (1 MB)

  qk_kernel<<<Bb * Ll / 4, 256, 0, stream>>>(x, Wt, Wx, bh, qp, kp);
  xt_kernel<<<Bb * (Ll / 32) * (Dd / 32), 256, 0, stream>>>(x, xT);
  attn_kernel<<<Bb * (Ll / 16), 256, 0, stream>>>(xT, qp, kp, Wa, ba, out);
}
